// KANLayer_48550310314628
// MI455X (gfx1250) — matrix operand order repaired and checked by reading, compiled, not run on hardware
//
#include <hip/hip_runtime.h>
#include <hip/hip_bf16.h>

typedef __bf16 bf16;
typedef __bf16 bf16x8 __attribute__((ext_vector_type(8)));
typedef __bf16 v16bf  __attribute__((ext_vector_type(16)));
typedef float  v8f    __attribute__((ext_vector_type(8)));
typedef int    v4i    __attribute__((ext_vector_type(4)));

#define C_DIM 768
#define O_DIM 768
#define T_DIM 16384
#define NCOEF 8
#define K_DIM (C_DIM + C_DIM * NCOEF)   // 6912 = 768 (silu path) + 6144 (spline path)

#define BM 128
#define BN 128
#define BK 32
#define LDSK 40   // padded LDS row stride in bf16 elements (80 B -> conflict-free b128 reads)

// ---------------------------------------------------------------------------
// CDNA5 async global->LDS copy (16 B per lane), ASYNCcnt-tracked.
// ---------------------------------------------------------------------------
__device__ __forceinline__ void async_cp_b128(const void* g, void* l) {
#if __has_builtin(__builtin_amdgcn_global_load_async_to_lds_b128)
    __builtin_amdgcn_global_load_async_to_lds_b128(
        (__attribute__((address_space(1))) v4i*)(v4i*)(void*)(size_t)g,
        (__attribute__((address_space(3))) v4i*)(v4i*)l, 0, 0);
#else
    unsigned lofs = (unsigned)(size_t)(__attribute__((address_space(3))) char*)l;
    asm volatile("global_load_async_to_lds_b128 %0, %1, off"
                 :: "v"(lofs), "v"(g) : "memory");
#endif
}

__device__ __forceinline__ void wait_async0() {
#if __has_builtin(__builtin_amdgcn_s_wait_asynccnt)
    __builtin_amdgcn_s_wait_asynccnt(0);
#else
    asm volatile("s_wait_asynccnt 0x0" ::: "memory");
#endif
}

// ---------------------------------------------------------------------------
// Pass 1: pack W[o][k] bf16, k-major: [ base_weight(768) | spline_weight*scaler (6144) ]
// ---------------------------------------------------------------------------
__global__ __launch_bounds__(256)
void kan_prep_weights(const float* __restrict__ bw,
                      const float* __restrict__ sw,
                      const float* __restrict__ sc,
                      bf16* __restrict__ W) {
    const int id = blockIdx.x * blockDim.x + threadIdx.x;   // id = o*C + c
    const int o = id / C_DIM;
    const int c = id - o * C_DIM;

    const float s = sc[id];
    W[(size_t)o * K_DIM + c] = (bf16)bw[id];

    const float* swp = sw + (size_t)id * NCOEF;
    bf16x8 v;
#pragma unroll
    for (int g = 0; g < NCOEF; ++g) v[g] = (bf16)(swp[g] * s);
    *(bf16x8*)(W + (size_t)o * K_DIM + C_DIM + (size_t)c * NCOEF) = v;
}

// ---------------------------------------------------------------------------
// Pass 2: pack A[t][k] bf16: [ silu(x)(768) | b-spline bases (768*8) ]
// Degree-3 Cox–de Boor on uniform knots g_j = -2.2 + 0.4*j, j=0..11
// ---------------------------------------------------------------------------
__global__ __launch_bounds__(256)
void kan_prep_act(const float* __restrict__ x, bf16* __restrict__ A) {
    const int id = blockIdx.x * blockDim.x + threadIdx.x;   // id = t*C + c
    const int t = id / C_DIM;
    const int c = id - t * C_DIM;

    const float xv = x[id];
    const float silu = xv / (1.0f + __expf(-xv));

    const float h = 0.4f;
    const float g0 = -2.2f;
    const float inv_h  = 2.5f;
    const float inv_2h = 1.25f;
    const float inv_3h = 0.8333333333f;

    float b0[11], b1[10], b2[9];
#pragma unroll
    for (int j = 0; j < 11; ++j) {
        const float gj = g0 + h * (float)j;
        b0[j] = (xv >= gj && xv < gj + h) ? 1.0f : 0.0f;
    }
#pragma unroll
    for (int j = 0; j < 10; ++j) {
        const float gj = g0 + h * (float)j;
        b1[j] = (xv - gj) * inv_h * b0[j] + (gj + 2.0f * h - xv) * inv_h * b0[j + 1];
    }
#pragma unroll
    for (int j = 0; j < 9; ++j) {
        const float gj = g0 + h * (float)j;
        b2[j] = (xv - gj) * inv_2h * b1[j] + (gj + 3.0f * h - xv) * inv_2h * b1[j + 1];
    }
    bf16x8 b3;
#pragma unroll
    for (int j = 0; j < 8; ++j) {
        const float gj = g0 + h * (float)j;
        b3[j] = (bf16)((xv - gj) * inv_3h * b2[j] + (gj + 4.0f * h - xv) * inv_3h * b2[j + 1]);
    }

    A[(size_t)t * K_DIM + c] = (bf16)silu;
    *(bf16x8*)(A + (size_t)t * K_DIM + C_DIM + (size_t)c * NCOEF) = b3;
}

// ---------------------------------------------------------------------------
// Pass 3: out[T][O] = A[T][K] * W[O][K]^T, bf16 WMMA, fp32 accumulate.
// WG tile 128x128, 8 waves as 4(M) x 2(N); wave tile 32x64 = 2x4 WMMA frags.
// Double-buffered LDS (explicit pointer swap, no unroll) filled by async DMA.
// ---------------------------------------------------------------------------
__global__ __launch_bounds__(256)
void kan_gemm(const bf16* __restrict__ A, const bf16* __restrict__ W,
              float* __restrict__ out) {
    __shared__ __align__(16) bf16 lA[2][BM * LDSK];
    __shared__ __align__(16) bf16 lB[2][BN * LDSK];

    const int tid   = threadIdx.x;
    const int lane  = tid & 31;
    const int wave  = tid >> 5;
    const int waveM = wave & 3;      // 0..3
    const int waveN = wave >> 2;     // 0..1
    const int tileM = blockIdx.x * BM;
    const int tileN = blockIdx.y * BN;

    // Global->LDS staging: each 32-elem (64 B) row = 4 chunks of 8 bf16 (16 B).
    const int chunk = tid & 3;
    const int srow  = tid >> 2;      // 0..63

    const int halfSel = lane >> 4;   // 0: lanes 0-15, 1: lanes 16-31
    const int lrow    = lane & 15;

    // per-thread source row pointers (advance by 64 B per K-step)
    const bf16* gA0 = A + (size_t)(tileM + srow) * K_DIM + chunk * 8;
    const bf16* gA1 = gA0 + (size_t)64 * K_DIM;
    const bf16* gB0 = W + (size_t)(tileN + srow) * K_DIM + chunk * 8;
    const bf16* gB1 = gB0 + (size_t)64 * K_DIM;

    // per-thread LDS destinations (relative to a buffer base)
    const int dofs0 = srow * LDSK + chunk * 8;
    const int dofs1 = (srow + 64) * LDSK + chunk * 8;

    // fragment source offsets (relative to a buffer base)
    const int faRow0 = (waveM * 32 + lrow) * LDSK + halfSel * 8;        // mi=0
    const int faRow1 = (waveM * 32 + 16 + lrow) * LDSK + halfSel * 8;   // mi=1
    const int fbBase = (waveN * 64 + lrow) * LDSK + halfSel * 8;        // ni=0 (+16*LDSK per ni)

    v8f acc[2][4];
#pragma unroll
    for (int mi = 0; mi < 2; ++mi)
#pragma unroll
        for (int ni = 0; ni < 4; ++ni) acc[mi][ni] = v8f{};

    const int nk = K_DIM / BK;       // 216

    // double-buffer pointers (swapped each iteration; loop NOT unrolled)
    const bf16* rdA = &lA[0][0];
    const bf16* rdB = &lB[0][0];
    bf16*       wrA = &lA[1][0];
    bf16*       wrB = &lB[1][0];

    // preload tile 0 into the read buffer via async DMA
    async_cp_b128(gA0, (void*)(rdA + dofs0));
    async_cp_b128(gA1, (void*)(rdA + dofs1));
    async_cp_b128(gB0, (void*)(rdB + dofs0));
    async_cp_b128(gB1, (void*)(rdB + dofs1));
    wait_async0();
    __syncthreads();

    union Frag { v16bf v; bf16x8 h[2]; };

    auto compute_step = [&](const bf16* bA, const bf16* bB) {
        // A fragments (16-bit A 16x32 ISA layout):
        //   lanes 0-15 : row M=lane,    K{0..7} + K{16..23}
        //   lanes 16-31: row M=lane-16, K{8..15} + K{24..31}
        Frag fa[2];
        fa[0].h[0] = *(const bf16x8*)(bA + faRow0);
        fa[0].h[1] = *(const bf16x8*)(bA + faRow0 + 16);
        fa[1].h[0] = *(const bf16x8*)(bA + faRow1);
        fa[1].h[1] = *(const bf16x8*)(bA + faRow1 + 16);
        // B fragments (32x16): lanes 0-15 col N=lane K{0..7}+K{16..23}; lanes 16-31 K{8..15}+K{24..31}
        Frag fb[4];
#pragma unroll
        for (int ni = 0; ni < 4; ++ni) {
            fb[ni].h[0] = *(const bf16x8*)(bB + fbBase + ni * 16 * LDSK);
            fb[ni].h[1] = *(const bf16x8*)(bB + fbBase + ni * 16 * LDSK + 16);
        }
#pragma unroll
        for (int mi = 0; mi < 2; ++mi)
#pragma unroll
            for (int ni = 0; ni < 4; ++ni)
                acc[mi][ni] = __builtin_amdgcn_wmma_f32_16x16x32_bf16(
                    false, fa[mi].v, false, fb[ni].v,
                    (short)0, acc[mi][ni], false, false);
    };

#pragma clang loop unroll(disable)
    for (int kt = 0; kt < nk - 1; ++kt) {
        // kick off DMA for the next K-slice into the write buffer
        const size_t koff = (size_t)(kt + 1) * BK;
        async_cp_b128(gA0 + koff, wrA + dofs0);
        async_cp_b128(gA1 + koff, wrA + dofs1);
        async_cp_b128(gB0 + koff, wrB + dofs0);
        async_cp_b128(gB1 + koff, wrB + dofs1);

        compute_step(rdA, rdB);

        // my async stores landed; barrier makes it workgroup-wide and also
        // guarantees everyone is done reading the read buffer before refill
        wait_async0();
        __syncthreads();

        // swap read/write buffers
        const bf16* tA = rdA; rdA = wrA; wrA = (bf16*)tA;
        const bf16* tB = rdB; rdB = wrB; wrB = (bf16*)tB;
    }
    // final K-slice (no prefetch)
    compute_step(rdA, rdB);

    // C/D f32 16x16 layout: VGPR r -> M = r + 8*halfSel, N = lane&15
#pragma unroll
    for (int mi = 0; mi < 2; ++mi) {
#pragma unroll
        for (int ni = 0; ni < 4; ++ni) {
            const int col = tileN + waveN * 64 + ni * 16 + lrow;
#pragma unroll
            for (int r = 0; r < 8; ++r) {
                const int row = tileM + waveM * 32 + mi * 16 + halfSel * 8 + r;
                out[(size_t)row * O_DIM + col] = acc[mi][ni][r];
            }
        }
    }
}

// ---------------------------------------------------------------------------
extern "C" void kernel_launch(void* const* d_in, const int* in_sizes, int n_in,
                              void* d_out, int out_size, void* d_ws, size_t ws_size,
                              hipStream_t stream) {
    const float* x  = (const float*)d_in[0];   // (4,4096,768)
    const float* bw = (const float*)d_in[1];   // (768,768)
    const float* sw = (const float*)d_in[2];   // (768,768,8)
    const float* sc = (const float*)d_in[3];   // (768,768)
    float* out = (float*)d_out;                // (4,4096,768) fp32

    bf16* W = (bf16*)d_ws;                         // 768 * 6912 bf16
    bf16* A = W + (size_t)O_DIM * K_DIM;           // 16384 * 6912 bf16

    kan_prep_weights<<<(O_DIM * C_DIM) / 256, 256, 0, stream>>>(bw, sw, sc, W);
    kan_prep_act<<<(T_DIM * C_DIM) / 256, 256, 0, stream>>>(x, A);

    dim3 grid(T_DIM / BM, O_DIM / BN);   // 128 x 6
    kan_gemm<<<grid, 256, 0, stream>>>(A, W, out);
}
